// ImageGraphHyperbolicGATClassifier_52029233824308
// MI455X (gfx1250) — compile-verified, matrix operand order checked by reading, not run on hardware
//
#include <hip/hip_runtime.h>
#include <hip/hip_bf16.h>
#include <cstdint>

// Problem constants (match reference)
#define NND   10000      // nodes
#define EE    160000     // edges (before self loops)
#define EPE   170000     // edges + self loops
#define GGR   64         // graphs
#define HH    8          // heads (layer 1)
#define DD    128        // per-head dim
#define HD    1024       // H*D == IN
#define PPD   32         // poincare dim
#define CCL   7          // classes
#define NEGS  0.2f
#define EPSV  1e-5f
#define CK    128        // K-chunk staged to LDS per TDM op

typedef __attribute__((ext_vector_type(16))) __bf16 v16bf;
typedef __attribute__((ext_vector_type(8)))  __bf16 v8bf;
typedef __attribute__((ext_vector_type(8)))  float  v8f;
typedef __attribute__((ext_vector_type(4)))  unsigned uv4;
typedef __attribute__((ext_vector_type(8)))  unsigned uv8;

// ---------- helpers ----------
__device__ __forceinline__ __bf16 f2bf(float f) {
  unsigned u = __float_as_uint(f);
  unsigned r = (u + 0x7FFFu + ((u >> 16) & 1u)) >> 16;   // RNE
  unsigned short s = (unsigned short)r;
  __bf16 o; __builtin_memcpy(&o, &s, 2); return o;
}
// ordered-uint encoding so unsigned atomicMax == float max
__device__ __forceinline__ unsigned encf(float f) {
  unsigned u = __float_as_uint(f);
  return (u & 0x80000000u) ? ~u : (u | 0x80000000u);
}
__device__ __forceinline__ float decf(unsigned e) {
  return (e & 0x80000000u) ? __uint_as_float(e & 0x7FFFFFFFu)
                           : __uint_as_float(~e);
}
__device__ __forceinline__ void get_edge(const int* __restrict__ ei, int e,
                                         int& s, int& d) {
  if (e < EE) { s = ei[e]; d = ei[EE + e]; }
  else        { s = e - EE; d = e - EE; }        // appended self loops
}

// ---------- init kernels ----------
__global__ void kcvt(__bf16* __restrict__ dst, const float* __restrict__ src, int n) {
  int i = blockIdx.x * blockDim.x + threadIdx.x;
  if (i < n) dst[i] = f2bf(src[i]);
}
__global__ void kzero(float* __restrict__ p, int n) {
  int i = blockIdx.x * blockDim.x + threadIdx.x;
  if (i < n) p[i] = 0.0f;
}
__global__ void kfill(unsigned* __restrict__ p, int n, unsigned v) {
  int i = blockIdx.x * blockDim.x + threadIdx.x;
  if (i < n) p[i] = v;
}

// ---------- WMMA GEMM: C[M,NC] = A[M,K] * B[K,NC], bf16 in / f32 out ----------
// Block = 8 waves covering one 16-row A stripe x 8 consecutive 16-col tiles.
// A chunk (16 x CK bf16 = 4KB) staged into LDS by the Tensor Data Mover
// (tensor_load_to_lds, TENSORcnt), then 4 WMMAs per stage per wave.
// NC is a compile-time constant so all B loads use immediate IOFFSETs off one
// base VGPR pair (no per-element 64-bit address math).
// M%16==0, NC%128==0, K%CK==0 (guaranteed by problem sizes).
template <int NC>
__global__ void kgemm_bf16(const __bf16* __restrict__ A, const __bf16* __restrict__ B,
                           float* __restrict__ C, int M, int K) {
  __shared__ __align__(16) __bf16 sA[16 * CK];

  int lane = threadIdx.x & 31;
  int wave = threadIdx.x >> 5;
  constexpr int tilesN = NC >> 4;
  constexpr int nblkN  = tilesN >> 3;          // 8 tn tiles per block
  int tm  = blockIdx.x / nblkN;
  int tnb = (blockIdx.x - tm * nblkN) * 8;
  int tn  = tnb + wave;
  int col  = tn * 16 + (lane & 15);
  int rowA = lane & 15;
  int kb   = (lane < 16) ? 0 : 8;              // 16-bit A/B lane K-base per ISA layout

  v8f acc = {0.f,0.f,0.f,0.f,0.f,0.f,0.f,0.f};

  // per-lane B base: row kb, column col; bumped by CK rows per staged chunk
  const __bf16* bptr = B + (size_t)kb * NC + col;

  for (int k0 = 0; k0 < K; k0 += CK) {
    if (wave == 0) {
      // ---- build Tensor DMA Descriptor (D#) and issue TDM load ----
      const __bf16* gA = A + (size_t)tm * 16 * K + k0;
      unsigned long long ga = (unsigned long long)(uintptr_t)gA;
      unsigned ldsa = (unsigned)(uintptr_t)(&sA[0]);   // LDS offset in low 32 bits
      uv4 g0;
      g0[0] = 1u;                                      // count=1, user mode
      g0[1] = ldsa;                                    // lds_addr
      g0[2] = (unsigned)ga;                            // global_addr[31:0]
      g0[3] = ((unsigned)(ga >> 32) & 0x01FFFFFFu) | 0x80000000u; // addr[56:32] | type=2
      uv8 g1;
      g1[0] = 1u << 16;                                // data_size = 2 bytes
      g1[1] = ((unsigned)K & 0xFFFFu) << 16;           // tensor_dim0[15:0]
      g1[2] = ((unsigned)K >> 16) | (((unsigned)M & 0xFFFFu) << 16); // dim0 hi | dim1 lo
      g1[3] = ((unsigned)M >> 16) | ((unsigned)CK << 16);           // dim1 hi | tile_dim0
      g1[4] = 16u;                                     // tile_dim1=16, tile_dim2=0
      g1[5] = (unsigned)K;                             // tensor_dim0_stride[31:0]
      g1[6] = 0u;
      g1[7] = 0u;
      uv4 g2 = {0u,0u,0u,0u};
      uv4 g3 = {0u,0u,0u,0u};
      asm volatile("tensor_load_to_lds %0, %1, %2, %3"
                   :: "s"(g0), "s"(g1), "s"(g2), "s"(g3) : "memory");
      __builtin_amdgcn_s_wait_tensorcnt(0);
    }
    __syncthreads();

#pragma unroll
    for (int kc = 0; kc < CK; kc += 32) {
      v8bf a0 = *reinterpret_cast<const v8bf*>(&sA[rowA * CK + kc + kb]);      // K=kb..kb+7
      v8bf a1 = *reinterpret_cast<const v8bf*>(&sA[rowA * CK + kc + kb + 16]); // K=kb+16..+23
      v16bf a, b;
#pragma unroll
      for (int i = 0; i < 8; ++i) { a[i] = a0[i]; a[i + 8] = a1[i]; }
#pragma unroll
      for (int i = 0; i < 8; ++i) {
        b[i]     = bptr[(kc + i) * NC];          // immediate offsets, same base
        b[i + 8] = bptr[(kc + 16 + i) * NC];
      }
      acc = __builtin_amdgcn_wmma_f32_16x16x32_bf16(
          false, a, false, b, (short)0, acc, false, false);
    }
    __syncthreads();
    bptr += (size_t)CK * NC;                     // one pointer bump per chunk
  }

  int rbase = tm * 16 + ((lane >> 4) << 3);      // M = r + 8*(lane/16)
  float* cp = C + (size_t)rbase * NC + col;
#pragma unroll
  for (int r = 0; r < 8; ++r)
    cp[(size_t)r * NC] = acc[r];
}

// ---------- attention coefficients: one wave per (node, head) ----------
__global__ void kalpha(const float* __restrict__ h, const float* __restrict__ asrc,
                       const float* __restrict__ adst, float* __restrict__ os,
                       float* __restrict__ od, int heads) {
  int gt = blockIdx.x * blockDim.x + threadIdx.x;
  int w = gt >> 5, lane = gt & 31;
  if (w >= NND * heads) return;
  int node = w / heads, hh = w - node * heads;
  const float* hp = h + (size_t)node * heads * DD + hh * DD;
  const float* as = asrc + hh * DD;
  const float* ad = adst + hh * DD;
  float ss = 0.f, sd = 0.f;
#pragma unroll
  for (int j = lane; j < DD; j += 32) { float v = hp[j]; ss += v * as[j]; sd += v * ad[j]; }
#pragma unroll
  for (int m = 16; m; m >>= 1) { ss += __shfl_xor(ss, m, 32); sd += __shfl_xor(sd, m, 32); }
  if (lane == 0) { os[w] = ss; od[w] = sd; }
}

// ---------- segment softmax pass 1: max per (dst, head) ----------
__global__ void kedge_max(const int* __restrict__ ei, const float* __restrict__ as,
                          const float* __restrict__ ad, unsigned* __restrict__ m,
                          int heads) {
  int idx = blockIdx.x * blockDim.x + threadIdx.x;
  if (idx >= EPE * heads) return;
  int e = idx / heads, hh = idx - e * heads;
  int s, d; get_edge(ei, e, s, d);
  float v = as[s * heads + hh] + ad[d * heads + hh];
  v = v > 0.f ? v : NEGS * v;
  atomicMax(&m[d * heads + hh], encf(v));
}

// ---------- segment softmax pass 2: exp + denominator ----------
__global__ void kedge_exp(const int* __restrict__ ei, const float* __restrict__ as,
                          const float* __restrict__ ad, const unsigned* __restrict__ m,
                          float* __restrict__ ex, float* __restrict__ den, int heads) {
  int idx = blockIdx.x * blockDim.x + threadIdx.x;
  if (idx >= EPE * heads) return;
  int e = idx / heads, hh = idx - e * heads;
  int s, d; get_edge(ei, e, s, d);
  float v = as[s * heads + hh] + ad[d * heads + hh];
  v = v > 0.f ? v : NEGS * v;
  float ev = __expf(v - decf(m[d * heads + hh]));
  ex[idx] = ev;
  atomicAdd(&den[d * heads + hh], ev);
}

// ---------- aggregation: out[dst] += h[src] * alpha. one block per edge ----------
__global__ void kagg(const int* __restrict__ ei, const float* __restrict__ h,
                     const float* __restrict__ ex, const float* __restrict__ den,
                     float* __restrict__ out, int heads) {
  int e = blockIdx.x;
  int s, d; get_edge(ei, e, s, d);
  __shared__ float w[HH];
  if ((int)threadIdx.x < heads)
    w[threadIdx.x] = ex[e * heads + threadIdx.x] / den[d * heads + threadIdx.x];
  __syncthreads();
  int f = threadIdx.x;                        // blockDim.x == DD == 128
  for (int hh = 0; hh < heads; ++hh)
    atomicAdd(&out[(size_t)d * heads * DD + hh * DD + f],
              h[(size_t)s * heads * DD + hh * DD + f] * w[hh]);
}

// ---------- layer1 finalize: +bias, ELU, re-encode bf16 for GEMM2 ----------
__global__ void kfin1(float* __restrict__ o, const float* __restrict__ b,
                      __bf16* __restrict__ ob, int total) {
  int i = blockIdx.x * blockDim.x + threadIdx.x;
  if (i >= total) return;
  float v = o[i] + b[i & (HD - 1)];
  v = v > 0.f ? v : __expf(v) - 1.f;
  o[i] = v;
  ob[i] = f2bf(v);
}

// ---------- layer2 finalize: +bias, ELU, pool-by-graph ----------
__global__ void kfin2(const float* __restrict__ o, const float* __restrict__ b,
                      const int* __restrict__ batch, float* __restrict__ pooled,
                      float* __restrict__ cnt, int total) {
  int i = blockIdx.x * blockDim.x + threadIdx.x;
  if (i >= total) return;
  int node = i / DD, f = i - node * DD;
  float v = o[i] + b[f];
  v = v > 0.f ? v : __expf(v) - 1.f;
  int g = batch[node];
  atomicAdd(&pooled[g * DD + f], v);
  if (f == 0) atomicAdd(&cnt[g], 1.0f);
}

// ---------- head: mean pool, Poincare clamp, classifier. single block ----------
__global__ void khead(const float* __restrict__ pooled, const float* __restrict__ cnt,
                      const float* __restrict__ Wp, const float* __restrict__ bp,
                      const float* __restrict__ Wc, const float* __restrict__ bc,
                      float* __restrict__ outp) {
  __shared__ float sp[GGR * DD];
  __shared__ float sz[GGR * PPD];
  int t = threadIdx.x;
  for (int i = t; i < GGR * DD; i += blockDim.x) {
    int g = i / DD;
    float c = cnt[g]; c = c > 1.f ? c : 1.f;
    sp[i] = pooled[i] / c;
  }
  __syncthreads();
  for (int o = t; o < GGR * PPD; o += blockDim.x) {
    int g = o / PPD, p = o - g * PPD;
    float acc = bp[p];
    for (int d2 = 0; d2 < DD; ++d2) acc += sp[g * DD + d2] * Wp[d2 * PPD + p];
    sz[o] = acc;
  }
  __syncthreads();
  if (t < GGR) {
    float s = 0.f;
    for (int p = 0; p < PPD; ++p) { float v = sz[t * PPD + p]; s += v * v; }
    float nrm = sqrtf(s);
    const float mx = 1.f - EPSV;
    if (nrm >= mx) { float f = mx / nrm; for (int p = 0; p < PPD; ++p) sz[t * PPD + p] *= f; }
  }
  __syncthreads();
  for (int o = t; o < GGR * CCL; o += blockDim.x) {       // logits first
    int g = o / CCL, c = o - g * CCL;
    float acc = bc[c];
    for (int p = 0; p < PPD; ++p) acc += sz[g * PPD + p] * Wc[p * CCL + c];
    outp[o] = acc;
  }
  for (int o = t; o < GGR * PPD; o += blockDim.x)          // then z
    outp[GGR * CCL + o] = sz[o];
}

#define CDIV(a, b) (((a) + (b) - 1) / (b))

extern "C" void kernel_launch(void* const* d_in, const int* in_sizes, int n_in,
                              void* d_out, int out_size, void* d_ws, size_t ws_size,
                              hipStream_t stream) {
  const float* x   = (const float*)d_in[0];
  const int*   ei  = (const int*)d_in[1];
  const int*   bat = (const int*)d_in[2];
  const float* W1  = (const float*)d_in[3];
  const float* as1 = (const float*)d_in[4];
  const float* ad1 = (const float*)d_in[5];
  const float* b1  = (const float*)d_in[6];
  const float* W2  = (const float*)d_in[7];
  const float* as2 = (const float*)d_in[8];
  const float* ad2 = (const float*)d_in[9];
  const float* b2  = (const float*)d_in[10];
  const float* Wp  = (const float*)d_in[11];
  const float* bp  = (const float*)d_in[12];
  const float* Wc  = (const float*)d_in[13];
  const float* bc  = (const float*)d_in[14];
  float* out = (float*)d_out;

  char* base = (char*)d_ws;
  size_t off = 0;
  auto alloc = [&](size_t bytes) -> void* {
    void* p = base + off;
    off = (off + bytes + 255) & ~(size_t)255;
    return p;
  };
  __bf16*  xb   = (__bf16*) alloc((size_t)NND * HD * 2);   // x in bf16; reused as h1a bf16
  __bf16*  W1b  = (__bf16*) alloc((size_t)HD * HD * 2);
  __bf16*  W2b  = (__bf16*) alloc((size_t)HD * DD * 2);
  float*   h1   = (float*)  alloc((size_t)NND * HD * 4);   // x@W1
  float*   o1   = (float*)  alloc((size_t)NND * HD * 4);   // layer1 aggregate -> h1 act
  float*   h2   = (float*)  alloc((size_t)NND * DD * 4);   // h1a@W2
  float*   o2   = (float*)  alloc((size_t)NND * DD * 4);   // layer2 aggregate
  float*   s1   = (float*)  alloc((size_t)NND * HH * 4);
  float*   d1   = (float*)  alloc((size_t)NND * HH * 4);
  unsigned* m1  = (unsigned*)alloc((size_t)NND * HH * 4);
  float*   den1 = (float*)  alloc((size_t)NND * HH * 4);
  float*   ex1  = (float*)  alloc((size_t)EPE * HH * 4);
  float*   s2   = (float*)  alloc((size_t)NND * 4);
  float*   d2   = (float*)  alloc((size_t)NND * 4);
  unsigned* m2  = (unsigned*)alloc((size_t)NND * 4);
  float*   den2 = (float*)  alloc((size_t)NND * 4);
  float*   ex2  = (float*)  alloc((size_t)EPE * 4);
  float*   pooled = (float*)alloc((size_t)GGR * DD * 4);
  float*   cnt  = (float*)  alloc((size_t)GGR * 4);
  __bf16*  h1ab = xb;   // alias: x dead after GEMM1

  const int BT = 256;
  const unsigned NEG_INF_ENC = 0x007FFFFFu;   // encf(-inf)

  // -- init / convert --
  kcvt <<<CDIV(NND * HD, BT), BT, 0, stream>>>(xb,  x,  NND * HD);
  kcvt <<<CDIV(HD * HD, BT),  BT, 0, stream>>>(W1b, W1, HD * HD);
  kcvt <<<CDIV(HD * DD, BT),  BT, 0, stream>>>(W2b, W2, HD * DD);
  kzero<<<CDIV(NND * HD, BT), BT, 0, stream>>>(o1,   NND * HD);
  kzero<<<CDIV(NND * DD, BT), BT, 0, stream>>>(o2,   NND * DD);
  kzero<<<CDIV(NND * HH, BT), BT, 0, stream>>>(den1, NND * HH);
  kzero<<<CDIV(NND, BT),      BT, 0, stream>>>(den2, NND);
  kzero<<<CDIV(GGR * DD, BT), BT, 0, stream>>>(pooled, GGR * DD);
  kzero<<<1, BT, 0, stream>>>(cnt, GGR);
  kfill<<<CDIV(NND * HH, BT), BT, 0, stream>>>(m1, NND * HH, NEG_INF_ENC);
  kfill<<<CDIV(NND, BT),      BT, 0, stream>>>(m2, NND,      NEG_INF_ENC);

  // -- layer 1 --
  kgemm_bf16<HD><<<(NND / 16) * (HD / 16 / 8), BT, 0, stream>>>(xb, W1b, h1, NND, HD);
  kalpha<<<NND * HH * 32 / BT, BT, 0, stream>>>(h1, as1, ad1, s1, d1, HH);
  kedge_max<<<CDIV(EPE * HH, BT), BT, 0, stream>>>(ei, s1, d1, m1, HH);
  kedge_exp<<<CDIV(EPE * HH, BT), BT, 0, stream>>>(ei, s1, d1, m1, ex1, den1, HH);
  kagg<<<EPE, DD, 0, stream>>>(ei, h1, ex1, den1, o1, HH);
  kfin1<<<CDIV(NND * HD, BT), BT, 0, stream>>>(o1, b1, h1ab, NND * HD);

  // -- layer 2 --
  kgemm_bf16<DD><<<(NND / 16) * (DD / 16 / 8), BT, 0, stream>>>(h1ab, W2b, h2, NND, HD);
  kalpha<<<NND * 1 * 32 / BT, BT, 0, stream>>>(h2, as2, ad2, s2, d2, 1);
  kedge_max<<<CDIV(EPE, BT), BT, 0, stream>>>(ei, s2, d2, m2, 1);
  kedge_exp<<<CDIV(EPE, BT), BT, 0, stream>>>(ei, s2, d2, m2, ex2, den2, 1);
  kagg<<<EPE, DD, 0, stream>>>(ei, h2, ex2, den2, o2, 1);
  kfin2<<<CDIV(NND * DD, BT), BT, 0, stream>>>(o2, b2, bat, pooled, cnt, NND * DD);

  // -- pooled head --
  khead<<<1, BT, 0, stream>>>(pooled, cnt, Wp, bp, Wc, bc, out);
}
